// KnnCtsLoss2_1443109012316
// MI455X (gfx1250) — matrix operand order repaired and checked by reading, not compiled, not run on hardware
//
#include <hip/hip_runtime.h>
#include <hip/hip_bf16.h>
#include <math.h>

// ---------------------------------------------------------------------------
// KNN contrastive loss on MI455X (gfx1250, wave32, WMMA)
//   features: (8192, 128) f32  ->  scalar f32 loss
//
// sim = Xn·Xnᵀ tiled 16x16 with v_wmma_f32_16x16x32_bf16 (K=128 -> 4 WMMA).
// Per row we only need (total exp-sum, top-6 values): top-k indices are only
// used to exclude columns from neg_sum, and subtracting exp(v/T) of the top-6
// *values* is equivalent.
// ---------------------------------------------------------------------------

#define N_ROWS 8192
#define DIM    128
#define TILE   16
#define NTILES (N_ROWS / TILE)      // 512
#define WAVES  8                    // waves per block in main kernel
#define ITERS  (NTILES / WAVES)     // 64 column tiles per wave
#define SIGMA  5
#define INV_T  10.0f                // 1 / TEMPERATURE

typedef __attribute__((ext_vector_type(8)))  float  v8f;
typedef __attribute__((ext_vector_type(8)))  __bf16 v8bf;
typedef __attribute__((ext_vector_type(16))) __bf16 v16bf;

// ----------------------------- helpers -------------------------------------

__device__ __forceinline__ unsigned short f32_to_bf16_rne(float f) {
    union { float f; unsigned u; } a; a.f = f;
    unsigned u = a.u;
    unsigned r = (u + 0x7FFFu + ((u >> 16) & 1u)) >> 16;   // round-nearest-even
    return (unsigned short)r;
}

// maintain t[0..5] sorted descending; single bottom insertion + bubble
__device__ __forceinline__ void top6_insert(float (&t)[6], float v) {
    if (v > t[5]) {
        t[5] = v;
#pragma unroll
        for (int k = 5; k > 0; --k) {
            float a = t[k - 1], b = t[k];
            t[k - 1] = fmaxf(a, b);
            t[k]     = fminf(a, b);
        }
    }
}

__device__ __forceinline__ void wave_lds_fence_acqrel() {
    // Same-wave DS ops complete in order (DScnt); we only need to stop the
    // compiler from reordering across the LDS handoff between lanes.
    __builtin_amdgcn_fence(__ATOMIC_RELEASE, "wavefront");
    __builtin_amdgcn_wave_barrier();
    __builtin_amdgcn_fence(__ATOMIC_ACQUIRE, "wavefront");
}

// B fragment (32x16 bf16, ISA layout): lanes 0-15: N=lane, K chunks at +0 and +16;
// lanes 16-31: N=lane-16, K chunks at +8 and +24.  Rows of X load as contiguous 16B chunks.
__device__ __forceinline__ void load_btile(const __bf16* __restrict__ xn,
                                           int j0, int lane, v16bf (&B)[4]) {
    const int koffB = (lane < 16) ? 0 : 8;
    const __bf16* p = xn + (size_t)(j0 + (lane & 15)) * DIM + koffB;
#pragma unroll
    for (int kk = 0; kk < 4; ++kk) {
        v8bf lo = *(const v8bf*)(p + kk * 32);
        v8bf hi = *(const v8bf*)(p + kk * 32 + 16);
        B[kk] = __builtin_shufflevector(lo, hi, 0, 1, 2, 3, 4, 5, 6, 7,
                                                8, 9, 10, 11, 12, 13, 14, 15);
    }
}

__device__ __forceinline__ v8f mm4(const v16bf (&A)[4], const v16bf (&B)[4]) {
    v8f acc = {};
#pragma unroll
    for (int kk = 0; kk < 4; ++kk)
        acc = __builtin_amdgcn_wmma_f32_16x16x32_bf16(
            false, A[kk], false, B[kk], (short)0, acc, false, false);
    return acc;
}

// consume one 16x16 sim tile held in WMMA C layout
__device__ __forceinline__ void consume_tile(v8f acc, float* __restrict__ tile,
                                             int lane, float (&rs)[8],
                                             float (&top)[6]) {
    const int crow = (lane < 16) ? 0 : 8;   // C layout: M = v + crow
    const int ccol = lane & 15;

    // exp-sum in register layout; cross-lane reduction deferred to the end
#pragma unroll
    for (int v = 0; v < 8; ++v)
        rs[v] += __expf(acc[v] * INV_T);

    // top-6 needs row-major view: bounce through this wave's private LDS tile
#pragma unroll
    for (int v = 0; v < 8; ++v)
        tile[(v + crow) * TILE + ccol] = acc[v];
    wave_lds_fence_acqrel();

    const int r = lane >> 1, h = lane & 1;  // lane owns (row r, 8-col half h)
    const float4* tp = (const float4*)(tile + r * TILE + h * 8);
    float4 a = tp[0], b = tp[1];

    float mx = fmaxf(fmaxf(fmaxf(a.x, a.y), fmaxf(a.z, a.w)),
                     fmaxf(fmaxf(b.x, b.y), fmaxf(b.z, b.w)));
    if (mx > top[5]) {                      // rare after warm-up
        top6_insert(top, a.x); top6_insert(top, a.y);
        top6_insert(top, a.z); top6_insert(top, a.w);
        top6_insert(top, b.x); top6_insert(top, b.y);
        top6_insert(top, b.z); top6_insert(top, b.w);
    }
    __builtin_amdgcn_wave_barrier();        // keep next tile's stores after reads
}

// ------------------------- kernel 1: normalize ------------------------------
__global__ void knn_normalize_bf16(const float* __restrict__ x,
                                   unsigned short* __restrict__ xn) {
    const int row  = blockIdx.x * 8 + (threadIdx.x >> 5);
    const int lane = threadIdx.x & 31;

    const float4 v = ((const float4*)(x + (size_t)row * DIM))[lane];
    float ss = v.x * v.x + v.y * v.y + v.z * v.z + v.w * v.w;
#pragma unroll
    for (int m = 1; m < 32; m <<= 1) ss += __shfl_xor(ss, m);

    const float inv = 1.0f / fmaxf(sqrtf(ss), 1e-12f);

    unsigned short b0 = f32_to_bf16_rne(v.x * inv);
    unsigned short b1 = f32_to_bf16_rne(v.y * inv);
    unsigned short b2 = f32_to_bf16_rne(v.z * inv);
    unsigned short b3 = f32_to_bf16_rne(v.w * inv);
    uint2 o;
    o.x = (unsigned)b0 | ((unsigned)b1 << 16);
    o.y = (unsigned)b2 | ((unsigned)b3 << 16);
    *((uint2*)(xn + (size_t)row * DIM + lane * 4)) = o;
}

// ------------------------- kernel 2: main -----------------------------------
__global__ void __launch_bounds__(WAVES * 32)
knn_sim_rowstats(const unsigned short* __restrict__ xnraw,
                 float* __restrict__ rowLoss) {
    __shared__ float tileLds[WAVES][TILE][TILE];   // per-wave private 16x16 tile
    __shared__ float sumsLds[WAVES][TILE];         // per-wave row exp-sums
    __shared__ float topLds[WAVES][32][6];         // per-wave per-lane top-6

    const __bf16* xn = (const __bf16*)xnraw;
    const int wave = threadIdx.x >> 5;
    const int lane = threadIdx.x & 31;
    const int i0   = blockIdx.x * TILE;
    float* tile = &tileLds[wave][0][0];

    // ---- A fragments (rows i0..i0+15), ISA 16-bit A layout:
    // lanes 0-15 : M=lane,    K chunks at +0 and +16 within each 32-block
    // lanes 16-31: M=lane-16, K chunks at +8 and +24
    const int mrow  = i0 + (lane & 15);
    const int koffA = (lane < 16) ? 0 : 8;
    v16bf A[4];
#pragma unroll
    for (int kk = 0; kk < 4; ++kk) {
        const __bf16* p = xn + (size_t)mrow * DIM + kk * 32 + koffA;
        v8bf lo = *(const v8bf*)(p);
        v8bf hi = *(const v8bf*)(p + 16);
        A[kk] = __builtin_shufflevector(lo, hi, 0, 1, 2, 3, 4, 5, 6, 7,
                                                8, 9, 10, 11, 12, 13, 14, 15);
    }

    // ---- per-lane state
    float rs[8];
#pragma unroll
    for (int v = 0; v < 8; ++v) rs[v] = 0.0f;
    float top[6];
#pragma unroll
    for (int k = 0; k < 6; ++k) top[k] = -1e30f;

    // ---- software-pipelined loop over this wave's column tiles
    v16bf B0[4], B1[4];
    load_btile(xn, wave * TILE, lane, B0);

    for (int it = 0; it < ITERS; it += 2) {
        const int jt0 = wave + it * WAVES;

        if (it + 4 < ITERS)   // global_prefetch_b8 a few tiles ahead
            __builtin_prefetch(
                (const void*)(xn + (size_t)((wave + (it + 4) * WAVES) * TILE +
                                            (lane & 15)) * DIM), 0, 1);

        if (it + 1 < ITERS) load_btile(xn, (jt0 + WAVES) * TILE, lane, B1);
        consume_tile(mm4(A, B0), tile, lane, rs, top);   // B1 loads in flight

        if (it + 2 < ITERS) load_btile(xn, (jt0 + 2 * WAVES) * TILE, lane, B0);
        if (it + 1 < ITERS)
            consume_tile(mm4(A, B1), tile, lane, rs, top);
    }

    // ---- fold exp-sums across the 16-lane groups (one butterfly, at the end)
#pragma unroll
    for (int v = 0; v < 8; ++v) {
#pragma unroll
        for (int m = 1; m < 16; m <<= 1) rs[v] += __shfl_xor(rs[v], m);
    }
    if (lane == 0) {
#pragma unroll
        for (int v = 0; v < 8; ++v) sumsLds[wave][v] = rs[v];       // rows 0-7
    }
    if (lane == 16) {
#pragma unroll
        for (int v = 0; v < 8; ++v) sumsLds[wave][8 + v] = rs[v];   // rows 8-15
    }
#pragma unroll
    for (int k = 0; k < 6; ++k) topLds[wave][lane][k] = top[k];
    __syncthreads();

    // ---- wave 0, lanes 0-15: finalize one row each
    if (wave == 0 && lane < 16) {
        float total = 0.0f;
        float t6[6];
#pragma unroll
        for (int k = 0; k < 6; ++k) t6[k] = -1e30f;

        for (int w = 0; w < WAVES; ++w) {
            total += sumsLds[w][lane];
#pragma unroll
            for (int hh = 0; hh < 2; ++hh) {
                const float* st = topLds[w][lane * 2 + hh];
#pragma unroll
                for (int k = 0; k < 6; ++k) top6_insert(t6, st[k]);
            }
        }

        // neg_sum = total - sum_{top6} exp(v/T);  positives exclude self (t6[0])
        float neg = total;
#pragma unroll
        for (int k = 0; k < 6; ++k) neg -= __expf(t6[k] * INV_T);
        neg = fmaxf(neg, 1e-20f);

        float posSum = 0.0f;
#pragma unroll
        for (int k = 1; k <= SIGMA; ++k) posSum += t6[k] * INV_T;

        rowLoss[i0 + lane] = (posSum - (float)SIGMA * __logf(neg)) / (float)SIGMA;
    }
}

// ------------------------- kernel 3: final reduce ---------------------------
__global__ void knn_final_reduce(const float* __restrict__ rowLoss,
                                 float* __restrict__ out) {
    __shared__ float partial[8];
    float s = 0.0f;
    for (int i = threadIdx.x; i < N_ROWS; i += 256) s += rowLoss[i];
#pragma unroll
    for (int m = 1; m < 32; m <<= 1) s += __shfl_xor(s, m);
    if ((threadIdx.x & 31) == 0) partial[threadIdx.x >> 5] = s;
    __syncthreads();
    if (threadIdx.x == 0) {
        float t = 0.0f;
#pragma unroll
        for (int w = 0; w < 8; ++w) t += partial[w];
        out[0] = fmaxf(-(t / (float)N_ROWS), 0.0f);
    }
}

// ---------------------------------------------------------------------------

extern "C" void kernel_launch(void* const* d_in, const int* in_sizes, int n_in,
                              void* d_out, int out_size, void* d_ws, size_t ws_size,
                              hipStream_t stream) {
    const float* features = (const float*)d_in[0];

    // workspace: [0, 2MB) bf16 normalized features; then 8192 f32 row losses
    unsigned short* xn = (unsigned short*)d_ws;
    float* rowLoss = (float*)((char*)d_ws + (size_t)N_ROWS * DIM * sizeof(unsigned short));

    knn_normalize_bf16<<<N_ROWS / 8, 256, 0, stream>>>(features, xn);
    knn_sim_rowstats<<<NTILES, WAVES * 32, 0, stream>>>(xn, rowLoss);
    knn_final_reduce<<<1, 256, 0, stream>>>(rowLoss, (float*)d_out);
}
